// DotProductAttention_64939905516378
// MI455X (gfx1250) — compile-verified, matrix operand order checked by reading, not run on hardware
//
#include <hip/hip_runtime.h>

// ---------------------------------------------------------------------------
// Decomposed rel-pos attention for MI455X (gfx1250), wave32 + WMMA f16.
// Phase 1a: bias tables qh,qw (L x 32), kh,kw (32 x L) per (b,n)      -> ws
// Phase 1b: K,V pre-swizzled to f16 WMMA B-operand tiles (8KB/tile)   -> ws
// Phase 2 : flash attention; K/V tiles streamed to LDS with
//           GLOBAL_LOAD_ASYNC_TO_LDS_B128 (double-buffered, ASYNCcnt),
//           v_wmma_f32_16x16x32_f16 for QK^T and P*V.
// ws requirement: 16 MB bias (fp32) + 8 MB K/V f16 tiles = 24 MB.
// ---------------------------------------------------------------------------

typedef __attribute__((ext_vector_type(16))) _Float16 v16h;
typedef __attribute__((ext_vector_type(8)))  float    v8f;

#define LQ   1024
#define CD   64
// fp32 workspace offsets (floats)
#define QH_OFF 0
#define QW_OFF (32*1024*32)
#define KH_OFF (2*32*1024*32)
#define KW_OFF (3*32*1024*32)
#define WS_F32_FLOATS (4*32*1024*32)          // 4M floats = 16 MB
// f16 workspace offsets (halves), relative to wsh = (ws + WS_F32_FLOATS)
#define WSK_HOFF 0
#define WSV_HOFF (32*16*4096)                 // 2M halves = 4 MB

// A/B operand K-index map (16-bit WMMA): element e, lane half h ->
// K = (e&7) + 8*h + 16*(e>>3)
__device__ __forceinline__ int kmap(int e, int half) {
  return (e & 7) + 8 * half + 16 * (e >> 3);
}

// --------------------------- Phase 1a: bias tables --------------------------
__global__ __launch_bounds__(256) void bias_kernel(
    const float* __restrict__ Qg, const float* __restrict__ Kg,
    const float* __restrict__ rhq, const float* __restrict__ rwq,
    const float* __restrict__ rhk, const float* __restrict__ rwk,
    float* __restrict__ ws)
{
  int gid = blockIdx.x * 256 + threadIdx.x;     // 0 .. 4*2^20-1
  int arr = gid >> 20;
  int r   = gid & ((1 << 20) - 1);
  int bn  = r >> 15;
  int rem = r & 32767;

  const float *row, *rel;
  float *out;
  if (arr < 2) {
    int q = rem >> 5, j = rem & 31;
    row = Qg + ((size_t)bn * LQ + q) * CD;
    int d = (arr == 0 ? (q >> 5) : (q & 31)) - j + 31;   // in [0,62]
    rel = (arr == 0 ? rhq : rwq) + d * CD;
    out = ws + (arr == 0 ? QH_OFF : QW_OFF) + (size_t)bn * 32768 + rem;
  } else {
    int j = rem >> 10, k = rem & 1023;
    row = Kg + ((size_t)bn * LQ + k) * CD;
    int d = j - (arr == 2 ? (k >> 5) : (k & 31)) + 31;
    rel = (arr == 2 ? rhk : rwk) + d * CD;
    out = ws + (arr == 2 ? KH_OFF : KW_OFF) + (size_t)bn * 32768 + rem;
  }
  float s = 0.f;
#pragma unroll
  for (int i = 0; i < 16; ++i) {
    float4 a = ((const float4*)row)[i];
    float4 b = ((const float4*)rel)[i];
    s += a.x * b.x + a.y * b.y + a.z * b.z + a.w * b.w;
  }
  *out = s;
}

// ------------------- Phase 1b: K/V -> f16 B-operand tiles -------------------
__global__ __launch_bounds__(256) void prepack_kernel(
    const float* __restrict__ Kg, const float* __restrict__ Vg,
    _Float16* __restrict__ wsh)
{
  int gid   = blockIdx.x * 256 + threadIdx.x;   // 0 .. 2*2^21-1
  int which = gid >> 21;                        // 0 = K, 1 = V
  int r     = gid & ((1 << 21) - 1);
  int bnkt  = r >> 12;                          // bn*16 + kt
  int idx   = r & 4095;
  int e  = idx & 15;
  int ln = (idx >> 4) & 31;
  int op = idx >> 9;                            // operand 0..7
  int hf = ln >> 4, nn = ln & 15;
  int kv = kmap(e, hf);
  int kt = bnkt & 15, bn = bnkt >> 4;
  int key, c;
  if (which == 0) { key = kt * 64 + (op >> 1) * 16 + nn; c = (op & 1) * 32 + kv; }
  else            { key = kt * 64 + (op >> 2) * 32 + kv; c = (op & 3) * 16 + nn; }
  const float* src = (which == 0 ? Kg : Vg);
  float v = src[((size_t)bn * LQ + key) * CD + c];
  wsh[(which == 0 ? WSK_HOFF : WSV_HOFF) + (size_t)bnkt * 4096 + idx] = (_Float16)v;
}

// ------------------------------ Phase 2: flash ------------------------------
__device__ __forceinline__ void async_tile(const _Float16* __restrict__ gsrc,
                                           _Float16* ldst, int t) {
#pragma unroll
  for (int i = 0; i < 4; ++i) {
    unsigned lk = (unsigned)(uintptr_t)(ldst + t * 8 + i * 1024);  // LDS byte addr
    const _Float16* gs = gsrc + t * 8 + i * 1024;
    asm volatile("global_load_async_to_lds_b128 %0, %1, off"
                 :: "v"(lk), "v"(gs) : "memory");
  }
}

__global__ __launch_bounds__(128) void flash_kernel(
    const float* __restrict__ Qg, const float* __restrict__ ws,
    const _Float16* __restrict__ wsh, float* __restrict__ Out)
{
  // double-buffered K/V tiles already in B-operand layout (8KB each)
  __shared__ __attribute__((aligned(32))) _Float16 sK[2][4096];
  __shared__ __attribute__((aligned(32))) _Float16 sV[2][4096];
  // P in A-layout, wave-private: [wave][k_chunk(2)][lane(32)*e(16)]
  __shared__ __attribute__((aligned(32))) _Float16 sP[4][2][32 * 16];

  const int bn    = blockIdx.y;
  const int qtile = blockIdx.x;
  const int t     = threadIdx.x;
  const int wave  = t >> 5;
  const int lane  = t & 31;
  const int half  = lane >> 4;
  const int n     = lane & 15;
  const int qbase = qtile * 64 + wave * 16;

  const float* qh_p = ws + QH_OFF + (size_t)bn * 32768;
  const float* qw_p = ws + QW_OFF + (size_t)bn * 32768;
  const float* kh_p = ws + KH_OFF + (size_t)bn * 32768;
  const float* kw_p = ws + KW_OFF + (size_t)bn * 32768;
  const _Float16* gK = wsh + WSK_HOFF + (size_t)(bn * 16) * 4096;
  const _Float16* gV = wsh + WSV_HOFF + (size_t)(bn * 16) * 4096;

  // ---- this wave's 16 Q rows as two f16 A operands (C = 0..31, 32..63) ----
  v16h qA[2];
  {
    const float* qr = Qg + ((size_t)bn * LQ + (qbase + n)) * CD;  // A: M = lane&15
#pragma unroll
    for (int cc = 0; cc < 2; ++cc)
#pragma unroll
      for (int e = 0; e < 16; ++e)
        qA[cc][e] = (_Float16)qr[kmap(e, half) + 32 * cc];
  }

  v8f Oacc[4];
  const v8f vzero = {0.f, 0.f, 0.f, 0.f, 0.f, 0.f, 0.f, 0.f};
#pragma unroll
  for (int cs = 0; cs < 4; ++cs) Oacc[cs] = vzero;
  float rm[8], rl[8];
#pragma unroll
  for (int i = 0; i < 8; ++i) { rm[i] = -3.0e38f; rl[i] = 0.f; }

  // prologue: start tile 0 into buffer 0
  async_tile(gK, &sK[0][0], t);
  async_tile(gV, &sV[0][0], t);

  for (int kt = 0; kt < 16; ++kt) {
    const int cur = kt & 1;
    const int k0  = kt * 64;

    asm volatile("s_wait_asynccnt 0x0" ::: "memory");  // my tile portion landed
    __syncthreads();                                   // whole tile visible

    if (kt + 1 < 16) {                                 // overlap next copy
      async_tile(gK + (size_t)(kt + 1) * 4096, &sK[cur ^ 1][0], t);
      async_tile(gV + (size_t)(kt + 1) * 4096, &sV[cur ^ 1][0], t);
    }
    const _Float16* sKb = &sK[cur][0];
    const _Float16* sVb = &sV[cur][0];

    // ---- scores: preload all 8 B operands, then 8 back-to-back WMMAs ----
    v16h kB[8];
#pragma unroll
    for (int op = 0; op < 8; ++op)
      kB[op] = *(const v16h*)(sKb + (op * 32 + lane) * 16);

    v8f S[4];
#pragma unroll
    for (int sub = 0; sub < 4; ++sub)
      S[sub] = __builtin_amdgcn_wmma_f32_16x16x32_f16(false, qA[0], false,
                                                      kB[sub * 2], (short)0,
                                                      vzero, false, false);
#pragma unroll
    for (int sub = 0; sub < 4; ++sub)
      S[sub] = __builtin_amdgcn_wmma_f32_16x16x32_f16(false, qA[1], false,
                                                      kB[sub * 2 + 1], (short)0,
                                                      S[sub], false, false);

    // ---- scale + rel-pos bias (C-layout: row i+8*half, col n) ----
#pragma unroll
    for (int sub = 0; sub < 4; ++sub)
#pragma unroll
      for (int i = 0; i < 8; ++i) {
        int q = qbase + i + 8 * half;
        int k = k0 + sub * 16 + n;
        float b = qh_p[q * 32 + (k >> 5)] + qw_p[q * 32 + (k & 31)]
                + kh_p[(q >> 5) * 1024 + k] + kw_p[(q & 31) * 1024 + k];
        S[sub][i] = S[sub][i] * 0.125f + b;    // 1/sqrt(64) on QK only
      }

    // ---- online softmax (reduce across the 16-lane column group) ----
#pragma unroll
    for (int i = 0; i < 8; ++i) {
      float tm = fmaxf(fmaxf(S[0][i], S[1][i]), fmaxf(S[2][i], S[3][i]));
      tm = fmaxf(tm, __shfl_xor(tm, 8));
      tm = fmaxf(tm, __shfl_xor(tm, 4));
      tm = fmaxf(tm, __shfl_xor(tm, 2));
      tm = fmaxf(tm, __shfl_xor(tm, 1));
      float newm = fmaxf(rm[i], tm);
      float corr = __expf(rm[i] - newm);
      rm[i] = newm;
      float ts = 0.f;
#pragma unroll
      for (int sub = 0; sub < 4; ++sub) {
        float p = __expf(S[sub][i] - newm);
        S[sub][i] = p;
        ts += p;
      }
      ts += __shfl_xor(ts, 8); ts += __shfl_xor(ts, 4);
      ts += __shfl_xor(ts, 2); ts += __shfl_xor(ts, 1);
      rl[i] = rl[i] * corr + ts;
#pragma unroll
      for (int cs = 0; cs < 4; ++cs) Oacc[cs][i] *= corr;
    }

    // ---- P: C-layout regs -> A-layout in wave-private LDS ----
#pragma unroll
    for (int sub = 0; sub < 4; ++sub)
#pragma unroll
      for (int i = 0; i < 8; ++i) {
        int mrow = i + 8 * half;
        int kl   = sub * 16 + n;
        int kc = kl >> 5, K32 = kl & 31;
        int hf = (K32 >> 3) & 1, e = (K32 & 7) + 8 * (K32 >> 4);
        sP[wave][kc][(hf * 16 + mrow) * 16 + e] = (_Float16)S[sub][i];
      }
    asm volatile("s_wait_dscnt 0x0" ::: "memory");  // per-wave LDS RAW fence

    // ---- O += P * V : preload operands, then 8 back-to-back WMMAs ----
    v16h aP0 = *(const v16h*)(sP[wave][0] + lane * 16);
    v16h aP1 = *(const v16h*)(sP[wave][1] + lane * 16);
    v16h vB[8];
#pragma unroll
    for (int op = 0; op < 8; ++op)
      vB[op] = *(const v16h*)(sVb + (op * 32 + lane) * 16);
#pragma unroll
    for (int cs = 0; cs < 4; ++cs)
      Oacc[cs] = __builtin_amdgcn_wmma_f32_16x16x32_f16(false, aP0, false,
                                                        vB[cs], (short)0,
                                                        Oacc[cs], false, false);
#pragma unroll
    for (int cs = 0; cs < 4; ++cs)
      Oacc[cs] = __builtin_amdgcn_wmma_f32_16x16x32_f16(false, aP1, false,
                                                        vB[4 + cs], (short)0,
                                                        Oacc[cs], false, false);

    __syncthreads();   // all waves done reading before buffer reuse
  }

  // ---- epilogue: normalize, add residual Q, store fp32 ----
#pragma unroll
  for (int cs = 0; cs < 4; ++cs)
#pragma unroll
    for (int i = 0; i < 8; ++i) {
      int q = qbase + i + 8 * half;
      int c = cs * 16 + n;
      size_t idx = ((size_t)bn * LQ + q) * CD + c;
      Out[idx] = Oacc[cs][i] / rl[i] + Qg[idx];
    }
}

// ---------------------------------------------------------------------------
extern "C" void kernel_launch(void* const* d_in, const int* in_sizes, int n_in,
                              void* d_out, int out_size, void* d_ws, size_t ws_size,
                              hipStream_t stream) {
  (void)in_sizes; (void)n_in; (void)out_size; (void)ws_size;
  const float* q   = (const float*)d_in[0];
  const float* k   = (const float*)d_in[1];
  const float* v   = (const float*)d_in[2];
  const float* rhq = (const float*)d_in[3];
  const float* rwq = (const float*)d_in[4];
  const float* rhk = (const float*)d_in[5];
  const float* rwk = (const float*)d_in[6];
  float*    ws  = (float*)d_ws;                       // needs 24 MB
  _Float16* wsh = (_Float16*)(ws + WS_F32_FLOATS);
  float*    out = (float*)d_out;

  bias_kernel<<<16384, 256, 0, stream>>>(q, k, rhq, rwq, rhk, rwk, ws);
  prepack_kernel<<<16384, 256, 0, stream>>>(k, v, wsh);
  flash_kernel<<<dim3(16, 32), 128, 0, stream>>>(q, ws, wsh, out);
}
